// MixedTraditionalMLP_82248623718485
// MI455X (gfx1250) — compile-verified, matrix operand order checked by reading, not run on hardware
//
#include <hip/hip_runtime.h>
#include <math.h>

typedef __attribute__((ext_vector_type(16))) _Float16 v16h;
typedef __attribute__((ext_vector_type(8)))  _Float16 v8h;
typedef __attribute__((ext_vector_type(8)))  float    v8f;
typedef __attribute__((ext_vector_type(4)))  _Float16 v4h;
typedef _Float16 half_t;

#define DM  4096
#define DFF 11008
#define RK  16
#define NT  8192

__constant__ float NF4_LUT[16] = {
  -1.0f, -0.6961928009986877f, -0.5250730514526367f, -0.39491748809814453f,
  -0.28444138169288635f, -0.18477343022823334f, -0.09105003625154495f, 0.0f,
  0.07958029955625534f, 0.16093020141124725f, 0.24611230194568634f, 0.33791524171829224f,
  0.44070982933044434f, 0.5626170039176941f, 0.7229568362236023f, 1.0f };

// ---------------------------------------------------------------------------
// NF4 dequant: int32 codes [N*K] + per-64-block absmax -> f16 weights
// ---------------------------------------------------------------------------
__global__ void dequant_nf4(const int* __restrict__ codes,
                            const float* __restrict__ absmax,
                            half_t* __restrict__ out, int total4) {
  int idx = blockIdx.x * blockDim.x + threadIdx.x;
  if (idx >= total4) return;
  const int4 c = ((const int4*)codes)[idx];
  const float s = absmax[(idx * 4) >> 6];
  v4h h;
  h[0] = (half_t)(NF4_LUT[c.x & 15] * s);
  h[1] = (half_t)(NF4_LUT[c.y & 15] * s);
  h[2] = (half_t)(NF4_LUT[c.z & 15] * s);
  h[3] = (half_t)(NF4_LUT[c.w & 15] * s);
  *(v4h*)(out + (size_t)idx * 4) = h;
}

__global__ void f32_to_f16(const float* __restrict__ in,
                           half_t* __restrict__ out, int total4) {
  int idx = blockIdx.x * blockDim.x + threadIdx.x;
  if (idx >= total4) return;
  const float4 v = ((const float4*)in)[idx];
  v4h h;
  h[0] = (half_t)v.x; h[1] = (half_t)v.y; h[2] = (half_t)v.z; h[3] = (half_t)v.w;
  *(v4h*)(out + (size_t)idx * 4) = h;
}

// ---------------------------------------------------------------------------
// LoRA first stage: T[M x 16] = X[M x K] @ A[K x 16]
// ---------------------------------------------------------------------------
__global__ void lora_t_f32(const float* __restrict__ X, const float* __restrict__ A,
                           float* __restrict__ T, int K) {
  int idx = blockIdx.x * blockDim.x + threadIdx.x;
  int m = idx >> 4, r = idx & 15;
  const float* x = X + (size_t)m * K;
  float s = 0.f;
  for (int k = 0; k < K; ++k) s += x[k] * A[(size_t)k * RK + r];
  T[idx] = s;
}

__global__ void lora_t_f16(const half_t* __restrict__ X, const float* __restrict__ A,
                           float* __restrict__ T, int K) {
  int idx = blockIdx.x * blockDim.x + threadIdx.x;
  int m = idx >> 4, r = idx & 15;
  const half_t* x = X + (size_t)m * K;
  float s = 0.f;
  for (int k = 0; k < K; ++k) s += (float)x[k] * A[(size_t)k * RK + r];
  T[idx] = s;
}

// ---------------------------------------------------------------------------
// WMMA fragment gather from LDS (CDNA5 16-bit A/B layout):
// lanes 0-15 hold K {0..7, 16..23}, lanes 16-31 hold K {8..15, 24..31}.
// Two explicit 16B vector loads -> exactly 2x ds_load_b128 per fragment.
// All address terms are multiples of 8 halves (LDT=72 -> 144B rows, ks*32 ->
// 64B, kb -> 0/16B, buffer stride 36864B), so 16B alignment is guaranteed.
// ---------------------------------------------------------------------------
__device__ inline v16h frag_ld(const half_t* p, int lane) {
  const int kb = (lane & 16) ? 8 : 0;
  const v8h lo = *(const v8h*)(p + kb);        // K group {0..7}+base
  const v8h hi = *(const v8h*)(p + 16 + kb);   // K group {16..23}+base
  return __builtin_shufflevector(lo, hi, 0, 1, 2, 3, 4, 5, 6, 7,
                                         8, 9, 10, 11, 12, 13, 14, 15);
}

// ---------------------------------------------------------------------------
// Y[M x N] = Xh[M x K] @ Wh[N x K]^T  + T[M x 16] @ Blora[16 x N]
// 256 threads = 8 wave32 (2x4); 128x128x64 tile; 4x2 WMMA tiles per wave.
// Double-buffered LDS filled by GLOBAL_LOAD_ASYNC_TO_LDS_B128 (ASYNCcnt),
// so the copy engine runs one K-tile ahead of the WMMA chain.
// ---------------------------------------------------------------------------
template <bool FUSE_SILU>
__global__ __launch_bounds__(256)
void gemm_wmma(const half_t* __restrict__ Xh, const half_t* __restrict__ Wh,
               const float* __restrict__ T, const float* __restrict__ Blora,
               float* __restrict__ Yf, half_t* __restrict__ Yh,
               int M, int N, int K) {
  constexpr int BM = 128, BN = 128, BK = 64, LDT = 72;
  constexpr int BUFH = BM * LDT + BN * LDT;       // halves per buffer (A then B)
  constexpr int BUFB = BUFH * 2;                  // bytes per buffer (36864)
  alignas(16) __shared__ half_t smem[2 * BUFH];   // 72 KB of the 320 KB WGP LDS

  const int tid  = threadIdx.x;
  const int lane = tid & 31;
  const int wave = tid >> 5;
  const int wm   = wave >> 2;   // 0..1 : 64-row strip of C
  const int wn   = wave & 3;    // 0..3 : 32-col strip of C
  const int m0   = blockIdx.y * BM;
  const int n0   = blockIdx.x * BN;

  // ---- per-lane async-copy geometry (loop invariant) --------------------
  // Each wave moves 16 rows of A and 16 rows of B per K-tile:
  // 4 instructions x 32 lanes x 16B per matrix.
  uint32_t voff[4], ldsA[4], ldsB[4];
#pragma unroll
  for (int t = 0; t < 4; ++t) {
    const int chunk = wave * 128 + t * 32 + lane; // 0..1023
    const int row   = chunk >> 3;                 // 0..127
    const int cc    = chunk & 7;                  // 16B chunk within 64-half row
    voff[t] = (uint32_t)(row * K + cc * 8) * 2u;  // byte offset vs SGPR base
    ldsA[t] = (uint32_t)(row * LDT + cc * 8) * 2u;
    ldsB[t] = (uint32_t)((BM * LDT) + row * LDT + cc * 8) * 2u;
  }
  const half_t* gA = Xh + (size_t)m0 * K;  // uniform SGPR base (GVS mode)
  const half_t* gB = Wh + (size_t)n0 * K;

  auto issue_tile = [&](int k0, uint32_t bufb) {
    const half_t* a = gA + k0;
    const half_t* b = gB + k0;
#pragma unroll
    for (int t = 0; t < 4; ++t) {
      asm volatile("global_load_async_to_lds_b128 %0, %1, %2"
                   :: "v"(ldsA[t] + bufb), "v"(voff[t]), "s"(a) : "memory");
      asm volatile("global_load_async_to_lds_b128 %0, %1, %2"
                   :: "v"(ldsB[t] + bufb), "v"(voff[t]), "s"(b) : "memory");
    }
  };

  const v8f vzero = {0.f, 0.f, 0.f, 0.f, 0.f, 0.f, 0.f, 0.f};
  v8f acc[4][2];
#pragma unroll
  for (int i = 0; i < 4; ++i)
#pragma unroll
    for (int j = 0; j < 2; ++j) acc[i][j] = vzero;

  // Prologue: start filling buffer 0.
  issue_tile(0, 0u);
  int cur = 0;

  for (int k0 = 0; k0 < K; k0 += BK) {
    if (k0 + BK < K) {
      // All waves finished reading buf[cur^1] last iteration (trailing
      // barrier), so it is safe to start overwriting it now.
      issue_tile(k0 + BK, cur ? 0u : (uint32_t)BUFB);
      // 16 outstanding, in-order completion: <=8 left means buf[cur] landed.
      asm volatile("s_wait_asynccnt 8" ::: "memory");
    } else {
      asm volatile("s_wait_asynccnt 0" ::: "memory");
    }
    __syncthreads();  // every wave's slice of buf[cur] is in LDS

    const half_t* sAc = smem + cur * BUFH;
    const half_t* sBc = sAc + BM * LDT;
#pragma unroll
    for (int ks = 0; ks < 2; ++ks) {
      v16h afrag[4], bfrag[2];
#pragma unroll
      for (int i = 0; i < 4; ++i)
        afrag[i] = frag_ld(&sAc[(wm * 64 + i * 16 + (lane & 15)) * LDT + ks * 32], lane);
#pragma unroll
      for (int j = 0; j < 2; ++j)
        bfrag[j] = frag_ld(&sBc[(wn * 32 + j * 16 + (lane & 15)) * LDT + ks * 32], lane);
#pragma unroll
      for (int i = 0; i < 4; ++i)
#pragma unroll
        for (int j = 0; j < 2; ++j)
          acc[i][j] = __builtin_amdgcn_wmma_f32_16x16x32_f16(
              false, afrag[i], false, bfrag[j], (short)0, acc[i][j], false, false);
    }
    __syncthreads();  // done reading buf[cur]; next iter may refill it
    cur ^= 1;
  }

  // Epilogue: rank-16 LoRA correction + (optional) SiLU + store.
  // C/D layout: element e maps to m_local = e + (lane>=16)*8, n = lane&15.
  const int lrow = (lane & 16) ? 8 : 0;
  const int lcol = lane & 15;
#pragma unroll
  for (int j = 0; j < 2; ++j) {
    const int n = n0 + wn * 32 + j * 16 + lcol;
    float bl[RK];
#pragma unroll
    for (int r = 0; r < RK; ++r) bl[r] = Blora[(size_t)r * N + n];
#pragma unroll
    for (int i = 0; i < 4; ++i) {
      const int mb = m0 + wm * 64 + i * 16 + lrow;
#pragma unroll
      for (int e = 0; e < 8; ++e) {
        const int m = mb + e;
        float v = acc[i][j][e];
        const float* t = T + (size_t)m * RK;
#pragma unroll
        for (int r = 0; r < RK; ++r) v += t[r] * bl[r];
        if (FUSE_SILU) {
          v = v * (1.0f / (1.0f + __expf(-v)));
          Yh[(size_t)m * N + n] = (half_t)v;
        } else {
          Yf[(size_t)m * N + n] = v;
        }
      }
    }
  }
}

// ---------------------------------------------------------------------------
extern "C" void kernel_launch(void* const* d_in, const int* in_sizes, int n_in,
                              void* d_out, int out_size, void* d_ws, size_t ws_size,
                              hipStream_t stream) {
  const float* x1  = (const float*)d_in[0];
  const int*   upC = (const int*)  d_in[1];
  const float* upS = (const float*)d_in[2];
  const float* upA = (const float*)d_in[3];
  const float* upB = (const float*)d_in[4];
  const int*   dnC = (const int*)  d_in[5];
  const float* dnS = (const float*)d_in[6];
  const float* dnA = (const float*)d_in[7];
  const float* dnB = (const float*)d_in[8];
  float* out = (float*)d_out;
  (void)in_sizes; (void)n_in; (void)out_size; (void)ws_size;

  char* ws = (char*)d_ws;
  size_t off = 0;
  auto take = [&](size_t bytes) -> void* {
    void* p = ws + off;
    off += (bytes + 255) & ~(size_t)255;
    return p;
  };
  half_t* x1h  = (half_t*)take((size_t)NT  * DM  * 2);  //  67 MB
  half_t* wUph = (half_t*)take((size_t)DFF * DM  * 2);  //  90 MB (fits L2)
  half_t* x2h  = (half_t*)take((size_t)NT  * DFF * 2);  // 180 MB
  half_t* wDnh = (half_t*)take((size_t)DM  * DFF * 2);  //  90 MB (fits L2)
  float*  tUp  = (float*) take((size_t)NT  * RK  * 4);
  float*  tDn  = (float*) take((size_t)NT  * RK  * 4);

  {
    int t4 = NT * DM / 4;
    f32_to_f16<<<(t4 + 255) / 256, 256, 0, stream>>>(x1, x1h, t4);
  }
  {
    int t4 = DFF * DM / 4;
    dequant_nf4<<<(t4 + 255) / 256, 256, 0, stream>>>(upC, upS, wUph, t4);
  }
  lora_t_f32<<<NT * RK / 256, 256, 0, stream>>>(x1, upA, tUp, DM);
  {
    dim3 g(DFF / 128, NT / 128);  // 86 x 64
    gemm_wmma<true><<<g, 256, 0, stream>>>(x1h, wUph, tUp, upB,
                                           nullptr, x2h, NT, DFF, DM);
  }
  {
    int t4 = DM * DFF / 4;
    dequant_nf4<<<(t4 + 255) / 256, 256, 0, stream>>>(dnC, dnS, wDnh, t4);
  }
  lora_t_f16<<<NT * RK / 256, 256, 0, stream>>>(x2h, dnA, tDn, DFF);
  {
    dim3 g(DM / 128, NT / 128);   // 32 x 64
    gemm_wmma<false><<<g, 256, 0, stream>>>(x2h, wDnh, tDn, dnB,
                                            out, nullptr, NT, DM, DFF);
  }
}
